// PIGNN_43293270344012
// MI455X (gfx1250) — compile-verified, matrix operand order checked by reading, not run on hardware
//
#include <hip/hip_runtime.h>

#define NPARAM 82
#define WAVES 8
#define N_TILES 65                        // 1KB f16 WMMA B-tiles staged in LDS
#define SMEM_BYTES (N_TILES * 1024 + WAVES * 6144)

typedef __attribute__((ext_vector_type(16))) _Float16 v16h;
typedef __attribute__((ext_vector_type(8)))  _Float16 v8h;
typedef __attribute__((ext_vector_type(8)))  float    v8f;

struct KParams { const float* p[NPARAM]; };

// ---------------- wave-local LDS ordering fence ----------------
__device__ __forceinline__ void lds_fence() {
  asm volatile("s_wait_dscnt 0x0" ::: "memory");
}

// 16-lane-half sum reduction via ds_swizzle (xor 1,2,4,8 stay within halves)
__device__ __forceinline__ float red16(float v) {
  v += __int_as_float(__builtin_amdgcn_ds_swizzle(__float_as_int(v), 0x041F));
  v += __int_as_float(__builtin_amdgcn_ds_swizzle(__float_as_int(v), 0x081F));
  v += __int_as_float(__builtin_amdgcn_ds_swizzle(__float_as_int(v), 0x101F));
  v += __int_as_float(__builtin_amdgcn_ds_swizzle(__float_as_int(v), 0x201F));
  return v;
}

__device__ __forceinline__ v16h cat16(v8h lo, v8h hi) {
  return __builtin_shufflevector(lo, hi, 0,1,2,3,4,5,6,7,8,9,10,11,12,13,14,15);
}

// A operand: plain row-major 16 x strideH f16 buffer -> WMMA A lane layout.
// lane m (<16): K = kBase+[0..7], kBase+[16..23]; lane m+16: +8 on both runs.
__device__ __forceinline__ v16h loadA(const _Float16* buf, int strideH, int kBase, int lane) {
  int m = lane & 15, b = lane >> 4;
  const _Float16* p = buf + m * strideH + kBase + b * 8;
  return cat16(*(const v8h*)p, *(const v8h*)(p + 16));
}

// B operand: pre-swizzled tile, 16 contiguous halves per lane.
__device__ __forceinline__ v16h loadB(const _Float16* sW, int tileIdx, int lane) {
  const _Float16* p = sW + tileIdx * 512 + lane * 16;
  return cat16(*(const v8h*)p, *(const v8h*)(p + 8));
}

// (16 x 32k) x (32k x 32) GEMM -> two f32 accumulators (N=0..15, N=16..31)
__device__ __forceinline__ void gemm2(v8f& d0, v8f& d1, const _Float16* A, int strideH,
                                      int nChunks, const _Float16* sW, int tBase, int lane) {
  v8f z = {0.f,0.f,0.f,0.f,0.f,0.f,0.f,0.f};
  d0 = z; d1 = z;
  for (int c = 0; c < nChunks; ++c) {
    v16h a  = loadA(A, strideH, c * 32, lane);
    v16h b0 = loadB(sW, tBase + c * 2 + 0, lane);
    v16h b1 = loadB(sW, tBase + c * 2 + 1, lane);
    d0 = __builtin_amdgcn_wmma_f32_16x16x32_f16(false, a, false, b0, (short)0, d0, false, false);
    d1 = __builtin_amdgcn_wmma_f32_16x16x32_f16(false, a, false, b1, (short)0, d1, false, false);
  }
}

// softplus(beta*x)/beta then LayerNorm(g,bb), done in D-matrix register layout.
// Moments form: the two cross-lane reduction trees (sum, sum-of-squares) are
// independent, so their ds_swizzle chains overlap.
__device__ __forceinline__ void act_ln(v8f& d0, v8f& d1, const float* b, const float* betaP,
                                       const float* g, const float* bb, int lane) {
  int c0 = lane & 15;
  float bias0 = b[c0], bias1 = b[c0 + 16];
  float g0 = g[c0], g1 = g[c0 + 16];
  float bb0 = bb[c0], bb1 = bb[c0 + 16];
  float beta = betaP[0];
  float inv = __builtin_amdgcn_rcpf(beta);   // single v_rcp_f32, uniform scalar
#pragma unroll
  for (int j = 0; j < 8; ++j) {
    float z0 = (d0[j] + bias0) * beta;
    float z1 = (d1[j] + bias1) * beta;
    float h0 = (fmaxf(z0, 0.f) + __logf(1.f + __expf(-fabsf(z0)))) * inv;
    float h1 = (fmaxf(z1, 0.f) + __logf(1.f + __expf(-fabsf(z1)))) * inv;
    float s  = red16(h0 + h1);               // independent chain 1
    float sq = red16(h0 * h0 + h1 * h1);     // independent chain 2
    float mu = s * 0.03125f;
    float var = sq * 0.03125f - mu * mu;
    float rs = __builtin_amdgcn_rsqf(var + 1e-5f);
    d0[j] = (h0 - mu) * rs * g0 + bb0;
    d1[j] = (h1 - mu) * rs * g1 + bb1;
  }
}

__device__ __forceinline__ void addBias(v8f& d0, v8f& d1, const float* b, float s, int lane) {
  int c0 = lane & 15;
  float b0 = b[c0] * s, b1 = b[c0 + 16] * s;
#pragma unroll
  for (int j = 0; j < 8; ++j) { d0[j] += b0; d1[j] += b1; }
}

// D registers -> plain row-major f16 buffer (row stride in halfs)
__device__ __forceinline__ void storeHalf(const v8f& d0, const v8f& d1, _Float16* dst,
                                          int stride, int colOff, int lane) {
  int c0 = lane & 15, rb = (lane >> 4) << 3;
#pragma unroll
  for (int j = 0; j < 8; ++j) {
    dst[(j + rb) * stride + colOff + c0]      = (_Float16)d0[j];
    dst[(j + rb) * stride + colOff + c0 + 16] = (_Float16)d1[j];
  }
}

// Stage a (K x Nact) row-major f32 weight matrix into zero-padded WMMA B-tiles.
__device__ void stageB(const float* W, int K, int Nact, int nChunks, int nNt,
                       _Float16* dst, int tid, int nthreads) {
  int total = nChunks * nNt * 512;
  for (int t = tid; t < total; t += nthreads) {
    int tile = t >> 9, e = t & 511;
    int lanev = e >> 4, h = e & 15;
    int n16 = lanev & 15, lb = lanev >> 4, hb = h >> 3;
    int k = (((hb << 1) | lb) << 3) | (h & 7);
    int chunk = tile / nNt, ntile = tile - chunk * nNt;
    int kk = chunk * 32 + k, nn = ntile * 16 + n16;
    float v = (kk < K && nn < Nact) ? W[kk * Nact + nn] : 0.f;
    dst[t] = (_Float16)v;
  }
}

__global__ void __launch_bounds__(256) pignn_kernel(KParams P, float* __restrict__ out, int Btot) {
  extern __shared__ char smemRaw[];
  _Float16* sW = (_Float16*)smemRaw;
  const int tid = threadIdx.x;

  // ---- stage all weights into LDS B-tile layout (once per block) ----
  stageB(P.p[7],  14, 32, 1, 2, sW + 0 * 512, tid, 256);   // enc W0
  stageB(P.p[12], 32, 32, 1, 2, sW + 2 * 512, tid, 256);   // enc W1
  for (int l = 0; l < 2; ++l) {
    int pb = 14 + l * 28, tb = 4 + l * 28;
    stageB(P.p[pb + 0],  42, 32, 2, 2, sW + (tb + 0)  * 512, tid, 256);  // buo W0
    stageB(P.p[pb + 5],  32, 32, 1, 2, sW + (tb + 4)  * 512, tid, 256);  // buo W1
    stageB(P.p[pb + 7],  44, 32, 2, 2, sW + (tb + 6)  * 512, tid, 256);  // hyd W0
    stageB(P.p[pb + 12], 32, 32, 1, 2, sW + (tb + 10) * 512, tid, 256);  // hyd W1
    stageB(P.p[pb + 14], 47, 32, 2, 2, sW + (tb + 12) * 512, tid, 256);  // thr W0
    stageB(P.p[pb + 19], 32, 32, 1, 2, sW + (tb + 16) * 512, tid, 256);  // thr W1
    stageB(P.p[pb + 21],128, 32, 4, 2, sW + (tb + 18) * 512, tid, 256);  // upd W0
    stageB(P.p[pb + 26], 32, 32, 1, 2, sW + (tb + 26) * 512, tid, 256);  // upd W1
  }
  stageB(P.p[70], 32, 32, 1, 2, sW + 60 * 512, tid, 256);  // ro W0
  stageB(P.p[75], 32, 32, 1, 2, sW + 62 * 512, tid, 256);  // ro W1
  stageB(P.p[80], 32,  9, 1, 1, sW + 64 * 512, tid, 256);  // ro W2 (N padded to 16)
  __syncthreads();

  const int lane = tid & 31, wid = tid >> 5;
  const int r0 = (blockIdx.x * WAVES + wid) * 16;
  if (r0 + 16 > Btot) return;

  _Float16* sIN   = sW + N_TILES * 512 + wid * 3072;  // 16x128 f16 input assembly
  _Float16* sHID  = sIN + 2048;                       // 16x32 f16 hidden
  _Float16* sHULL = sIN + 2560;                       // 16x32 f16 hull state
  const float* Z = P.p[0];

  v8f d0, d1;
  const v8f vzero = {0.f,0.f,0.f,0.f,0.f,0.f,0.f,0.f};

  // ---- encoder: 14 -> 32 -> 32 ----
  for (int t = lane; t < 512; t += 32) {
    int r = t >> 5, c = t & 31;
    sIN[t] = (c < 14) ? (_Float16)Z[(size_t)(r0 + r) * 14 + c] : (_Float16)0.f;
  }
  lds_fence();
  gemm2(d0, d1, sIN, 32, 1, sW, 0, lane);
  act_ln(d0, d1, P.p[8], P.p[9], P.p[10], P.p[11], lane);
  storeHalf(d0, d1, sHID, 32, 0, lane); lds_fence();
  gemm2(d0, d1, sHID, 32, 1, sW, 2, lane);
  addBias(d0, d1, P.p[13], 1.f, lane);
  storeHalf(d0, d1, sHULL, 32, 0, lane); lds_fence();

  // ---- 2 message-passing layers ----
  for (int l = 0; l < 2; ++l) {
    const int pb = 14 + l * 28, tb = 4 + l * 28;

    // thruster messages over 8 edges, summed in accumulators
    v8f aT0 = vzero, aT1 = vzero;
    const float* tx = P.p[1]; const float* te = P.p[2];
    for (int e = 0; e < 8; ++e) {
      for (int t = lane; t < 1024; t += 32) {
        int r = t >> 6, c = t & 63;
        _Float16 hv;
        if      (c < 8)  hv = (_Float16)tx[((size_t)(r0 + r) * 8 + e) * 8 + c];
        else if (c < 15) hv = (_Float16)te[((size_t)(r0 + r) * 8 + e) * 7 + (c - 8)];
        else if (c < 47) hv = sHULL[r * 32 + (c - 15)];
        else             hv = (_Float16)0.f;
        sIN[t] = hv;
      }
      lds_fence();
      gemm2(d0, d1, sIN, 64, 2, sW, tb + 12, lane);
      act_ln(d0, d1, P.p[pb + 15], P.p[pb + 16], P.p[pb + 17], P.p[pb + 18], lane);
      storeHalf(d0, d1, sHID, 32, 0, lane); lds_fence();
      gemm2(d0, d1, sHID, 32, 1, sW, tb + 16, lane);
      aT0 += d0; aT1 += d1;
    }
    addBias(aT0, aT1, P.p[pb + 20], 8.f, lane);  // bias summed over 8 edges

    // hydro MLP (44 -> 32 -> 32)
    {
      const float* hx = P.p[3]; const float* he = P.p[4];
      for (int t = lane; t < 1024; t += 32) {
        int r = t >> 6, c = t & 63;
        _Float16 hv;
        if      (c < 4)  hv = (_Float16)hx[(size_t)(r0 + r) * 4 + c];
        else if (c < 12) hv = (_Float16)he[(size_t)(r0 + r) * 8 + (c - 4)];
        else if (c < 44) hv = sHULL[r * 32 + (c - 12)];
        else             hv = (_Float16)0.f;
        sIN[t] = hv;
      }
      lds_fence();
      gemm2(d0, d1, sIN, 64, 2, sW, tb + 6, lane);
      act_ln(d0, d1, P.p[pb + 8], P.p[pb + 9], P.p[pb + 10], P.p[pb + 11], lane);
      storeHalf(d0, d1, sHID, 32, 0, lane); lds_fence();
      gemm2(d0, d1, sHID, 32, 1, sW, tb + 10, lane);
      addBias(d0, d1, P.p[pb + 13], 1.f, lane);
    }
    v8f aH0 = d0, aH1 = d1;

    // buoyancy MLP (42 -> 32 -> 32)
    {
      const float* bx = P.p[5]; const float* be = P.p[6];
      for (int t = lane; t < 1024; t += 32) {
        int r = t >> 6, c = t & 63;
        _Float16 hv;
        if      (c < 6)  hv = (_Float16)bx[(size_t)(r0 + r) * 6 + c];
        else if (c < 10) hv = (_Float16)be[(size_t)(r0 + r) * 4 + (c - 6)];
        else if (c < 42) hv = sHULL[r * 32 + (c - 10)];
        else             hv = (_Float16)0.f;
        sIN[t] = hv;
      }
      lds_fence();
      gemm2(d0, d1, sIN, 64, 2, sW, tb + 0, lane);
      act_ln(d0, d1, P.p[pb + 1], P.p[pb + 2], P.p[pb + 3], P.p[pb + 4], lane);
      storeHalf(d0, d1, sHID, 32, 0, lane); lds_fence();
      gemm2(d0, d1, sHID, 32, 1, sW, tb + 4, lane);
      addBias(d0, d1, P.p[pb + 6], 1.f, lane);
    }
    v8f aB0 = d0, aB1 = d1;

    // update MLP: [hull | agg_t | agg_h | agg_b] (128 -> 32 -> 32)
    for (int t = lane; t < 512; t += 32) {
      int r = t >> 5, c = t & 31;
      sIN[r * 128 + c] = sHULL[r * 32 + c];
    }
    storeHalf(aT0, aT1, sIN, 128, 32, lane);
    storeHalf(aH0, aH1, sIN, 128, 64, lane);
    storeHalf(aB0, aB1, sIN, 128, 96, lane);
    lds_fence();
    gemm2(d0, d1, sIN, 128, 4, sW, tb + 18, lane);
    act_ln(d0, d1, P.p[pb + 22], P.p[pb + 23], P.p[pb + 24], P.p[pb + 25], lane);
    storeHalf(d0, d1, sHID, 32, 0, lane); lds_fence();
    gemm2(d0, d1, sHID, 32, 1, sW, tb + 26, lane);
    addBias(d0, d1, P.p[pb + 27], 1.f, lane);
    storeHalf(d0, d1, sHULL, 32, 0, lane); lds_fence();
  }

  // ---- readout: 32 -> 32 -> 32 -> 9 ----
  gemm2(d0, d1, sHULL, 32, 1, sW, 60, lane);
  act_ln(d0, d1, P.p[71], P.p[72], P.p[73], P.p[74], lane);
  storeHalf(d0, d1, sHID, 32, 0, lane); lds_fence();
  gemm2(d0, d1, sHID, 32, 1, sW, 62, lane);
  act_ln(d0, d1, P.p[76], P.p[77], P.p[78], P.p[79], lane);
  storeHalf(d0, d1, sIN, 32, 0, lane); lds_fence();

  v8f f = vzero;
  {
    v16h a = loadA(sIN, 32, 0, lane);
    v16h b = loadB(sW, 64, lane);
    f = __builtin_amdgcn_wmma_f32_16x16x32_f16(false, a, false, b, (short)0, f, false, false);
  }
  float* fD = (float*)sHID;  // 16x16 f32 delta scratch
  {
    int c0 = lane & 15, rb = (lane >> 4) << 3;
    float b2v = (c0 < 9) ? P.p[81][c0] : 0.f;
#pragma unroll
    for (int j = 0; j < 8; ++j) fD[(j + rb) * 16 + c0] = f[j] + b2v;
  }
  lds_fence();

  // ---- trig epilogue + output ----
  for (int t = lane; t < 144; t += 32) {
    int r = t / 9, c = t - r * 9;
    const float* zr = Z + (size_t)(r0 + r) * 14;
    const float* dr = fD + r * 16;
    float val;
    if (c == 0) {
      float cp = dr[3] + zr[3], sp = dr[4] + zr[4];
      val = cp * dr[0] - sp * dr[1] + zr[0];
    } else if (c == 1) {
      float cp = dr[3] + zr[3], sp = dr[4] + zr[4];
      val = sp * dr[0] + cp * dr[1] + zr[1];
    } else {
      val = dr[c] + zr[c];
    }
    out[(size_t)(r0 + r) * 9 + c] = val;
  }
}

extern "C" void kernel_launch(void* const* d_in, const int* in_sizes, int n_in,
                              void* d_out, int out_size, void* d_ws, size_t ws_size,
                              hipStream_t stream) {
  (void)d_ws; (void)ws_size; (void)out_size;
  KParams P{};
  int n = n_in < NPARAM ? n_in : NPARAM;
  for (int i = 0; i < n; ++i) P.p[i] = (const float*)d_in[i];

  int B = in_sizes[0] / 14;                 // Z is (B, 14)
  int blocks = (B + (16 * WAVES - 1)) / (16 * WAVES);

  (void)hipFuncSetAttribute((const void*)pignn_kernel,
                            hipFuncAttributeMaxDynamicSharedMemorySize, SMEM_BYTES);
  pignn_kernel<<<blocks, 256, SMEM_BYTES, stream>>>(P, (float*)d_out, B);
}
